// EchoStateNetwork_23021024706867
// MI455X (gfx1250) — compile-verified, matrix operand order checked by reading, not run on hardware
//
#include <hip/hip_runtime.h>
#include <hip/hip_bf16.h>

typedef __attribute__((ext_vector_type(16))) _Float16 v16h;
typedef __attribute__((ext_vector_type(8)))  _Float16 v8h;
typedef __attribute__((ext_vector_type(8)))  float    v8f;

#define BB 32
#define TT 512
#define II 256
#define HH 1024
#define OO 256
#define BT (BB * TT)   // 16384

// ---------------- workspace layout (bytes) ----------------
#define OFF_XF16   ((size_t)0)                                  // BT*II f16   = 8 MB
#define OFF_WXHT   (OFF_XF16  + (size_t)BT * II * 2)            // HH*II f16   = 0.5 MB
#define OFF_WHHT   (OFF_WXHT  + (size_t)HH * II * 2)            // HH*HH f16   = 2 MB
#define OFF_WOUTT  (OFF_WHHT  + (size_t)HH * HH * 2)            // OO*HH f16   = 0.5 MB
#define OFF_XP     (OFF_WOUTT + (size_t)OO * HH * 2)            // T*B*H f32   = 64 MB
#define OFF_Z      (OFF_XP    + (size_t)TT * BB * HH * 4)       // BT*HH f16   = 32 MB
#define OFF_H0     (OFF_Z     + (size_t)BT * HH * 2)            // BB*HH f16
#define OFF_H1     (OFF_H0    + (size_t)BB * HH * 2)

union V16U { v16h v; v8h h[2]; };

// Fragment loader for 16-bit WMMA operands, (rows x K) row-major storage.
// Per CDNA5 ISA 7.12.2 (16-bit A 16x32): lane<16 holds K = {k0..k0+7, k0+16..k0+23},
// lane>=16 holds K = {k0+8..k0+15, k0+24..k0+31}; B (K x 16) loaded from the
// pre-transposed (N x K) array has the identical per-lane pattern.
__device__ __forceinline__ v16h load_frag(const _Float16* __restrict__ base,
                                          unsigned row, unsigned ld, unsigned k0,
                                          unsigned half) {
  const _Float16* p = base + (size_t)row * ld + k0 + half * 8u;
  V16U u;
  u.h[0] = *(const v8h*)(p);        // K = k0 + half*8 + 0..7
  u.h[1] = *(const v8h*)(p + 16);   // K = k0 + 16 + half*8 + 0..7
  return u.v;
}

// ---------------- elementwise helpers ----------------
__global__ void k_f32_to_f16(const float* __restrict__ in, _Float16* __restrict__ out, int n) {
  int i = blockIdx.x * blockDim.x + threadIdx.x;
  if (i < n) out[i] = (_Float16)in[i];
}

// src is R x C fp32 row-major; dst is C x R f16 row-major (transpose + convert)
__global__ void k_transpose_f16(const float* __restrict__ src, _Float16* __restrict__ dst,
                                int R, int C) {
  int i = blockIdx.x * blockDim.x + threadIdx.x;
  if (i < R * C) {
    int r = i / C, c = i % C;
    dst[(size_t)c * R + r] = (_Float16)src[i];
  }
}

__global__ void k_f16_to_f32(const _Float16* __restrict__ in, float* __restrict__ out, int n) {
  int i = blockIdx.x * blockDim.x + threadIdx.x;
  if (i < n) out[i] = (float)in[i];
}

// ---------------- phase 1: xp = x @ Wxh + b_rnn, stored (T,B,H) fp32 ----------------
__global__ void k_gemm_xp(const _Float16* __restrict__ A,   // BT x II (row-major)
                          const _Float16* __restrict__ Bt,  // HH x II (WxhT: [h][i])
                          const float* __restrict__ brnn,
                          float* __restrict__ xp) {
  unsigned tid  = threadIdx.x;
  unsigned lane = tid & 31u;
  unsigned wid  = blockIdx.x * (blockDim.x >> 5) + (tid >> 5);
  unsigned nt   = wid & 63u;       // 64 N-tiles (H/16)
  unsigned mt   = wid >> 6;        // 1024 M-tiles (BT/16)
  unsigned half = lane >> 4, r = lane & 15u;
  unsigned m0 = mt * 16u, n0 = nt * 16u;

  // two independent accumulators -> interleavable WMMAs (no D->A/B hazard NOPs)
  v8f c0 = {}, c1 = {};
#pragma unroll
  for (unsigned k0 = 0; k0 < II; k0 += 64) {
    v16h a0 = load_frag(A,  m0 + r, II, k0,       half);
    v16h b0 = load_frag(Bt, n0 + r, II, k0,       half);
    v16h a1 = load_frag(A,  m0 + r, II, k0 + 32u, half);
    v16h b1 = load_frag(Bt, n0 + r, II, k0 + 32u, half);
    c0 = __builtin_amdgcn_wmma_f32_16x16x32_f16(false, a0, false, b0, (short)0, c0, false, false);
    c1 = __builtin_amdgcn_wmma_f32_16x16x32_f16(false, a1, false, b1, (short)0, c1, false, false);
  }
  v8f c = c0 + c1;
  unsigned n = n0 + r;
  float bias = brnn[n];
#pragma unroll
  for (int v = 0; v < 8; ++v) {
    unsigned m = m0 + half * 8u + (unsigned)v;   // m = b*T + t
    unsigned b = m >> 9, t = m & 511u;           // T = 512 (power of two)
    xp[((size_t)t * BB + b) * HH + n] = c[v] + bias;
  }
}

// ---------------- phase 2: one recurrence step ----------------
// hnext = tanh(xp[t] + hprev @ Whh); also stores z[b*T+t] (f16)
__global__ void k_recur(const _Float16* __restrict__ hprev, // BB x HH f16
                        const _Float16* __restrict__ WhhT,  // HH x HH ([n][k])
                        const float* __restrict__ xp,       // (T,B,H) fp32
                        _Float16* __restrict__ hnext,
                        _Float16* __restrict__ z,           // BT x HH f16
                        int t) {
  unsigned tid  = threadIdx.x;
  unsigned lane = tid & 31u;
  unsigned wid  = blockIdx.x * (blockDim.x >> 5) + (tid >> 5);  // 0..127
  unsigned nt   = wid & 63u;       // 64 N-tiles
  unsigned mt   = wid >> 6;        // 2 M-tiles
  unsigned half = lane >> 4, r = lane & 15u;
  unsigned m0 = mt * 16u, n0 = nt * 16u;

  v8f c0 = {}, c1 = {};
#pragma unroll 4
  for (unsigned k0 = 0; k0 < HH; k0 += 64) {
    v16h a0 = load_frag(hprev, m0 + r, HH, k0,       half);
    v16h b0 = load_frag(WhhT,  n0 + r, HH, k0,       half);
    v16h a1 = load_frag(hprev, m0 + r, HH, k0 + 32u, half);
    v16h b1 = load_frag(WhhT,  n0 + r, HH, k0 + 32u, half);
    c0 = __builtin_amdgcn_wmma_f32_16x16x32_f16(false, a0, false, b0, (short)0, c0, false, false);
    c1 = __builtin_amdgcn_wmma_f32_16x16x32_f16(false, a1, false, b1, (short)0, c1, false, false);
  }
  v8f c = c0 + c1;
  unsigned n = n0 + r;
  const float* xprow = xp + (size_t)t * BB * HH;
#pragma unroll
  for (int v = 0; v < 8; ++v) {
    unsigned m = m0 + half * 8u + (unsigned)v;    // batch index b
    float hv = tanhf(c[v] + xprow[(size_t)m * HH + n]);
    _Float16 h16 = (_Float16)hv;
    hnext[(size_t)m * HH + n] = h16;
    z[((size_t)m * TT + t) * HH + n] = h16;
  }
}

// ---------------- phase 3: y = z @ W_out + b_out ----------------
__global__ void k_gemm_out(const _Float16* __restrict__ Z,    // BT x HH
                           const _Float16* __restrict__ WoT,  // OO x HH ([o][h])
                           const float* __restrict__ bout,
                           float* __restrict__ y) {            // BT x OO
  unsigned tid  = threadIdx.x;
  unsigned lane = tid & 31u;
  unsigned wid  = blockIdx.x * (blockDim.x >> 5) + (tid >> 5);
  unsigned nt   = wid & 15u;       // 16 N-tiles (O/16)
  unsigned mt   = wid >> 4;        // 1024 M-tiles
  unsigned half = lane >> 4, r = lane & 15u;
  unsigned m0 = mt * 16u, n0 = nt * 16u;

  v8f c0 = {}, c1 = {};
#pragma unroll 4
  for (unsigned k0 = 0; k0 < HH; k0 += 64) {
    v16h a0 = load_frag(Z,   m0 + r, HH, k0,       half);
    v16h b0 = load_frag(WoT, n0 + r, HH, k0,       half);
    v16h a1 = load_frag(Z,   m0 + r, HH, k0 + 32u, half);
    v16h b1 = load_frag(WoT, n0 + r, HH, k0 + 32u, half);
    c0 = __builtin_amdgcn_wmma_f32_16x16x32_f16(false, a0, false, b0, (short)0, c0, false, false);
    c1 = __builtin_amdgcn_wmma_f32_16x16x32_f16(false, a1, false, b1, (short)0, c1, false, false);
  }
  v8f c = c0 + c1;
  unsigned n = n0 + r;
  float bias = bout[n];
#pragma unroll
  for (int v = 0; v < 8; ++v) {
    unsigned m = m0 + half * 8u + (unsigned)v;
    y[(size_t)m * OO + n] = c[v] + bias;
  }
}

extern "C" void kernel_launch(void* const* d_in, const int* in_sizes, int n_in,
                              void* d_out, int out_size, void* d_ws, size_t ws_size,
                              hipStream_t stream) {
  (void)in_sizes; (void)n_in; (void)out_size; (void)ws_size;
  const float* x     = (const float*)d_in[0];   // (B,T,I)
  const float* h0    = (const float*)d_in[1];   // (B,H)
  const float* W_rnn = (const float*)d_in[2];   // (I+H, H)
  const float* b_rnn = (const float*)d_in[3];   // (H,)
  const float* W_out = (const float*)d_in[4];   // (H,O)
  const float* b_out = (const float*)d_in[5];   // (O,)

  char* ws = (char*)d_ws;
  _Float16* xf16  = (_Float16*)(ws + OFF_XF16);
  _Float16* WxhT  = (_Float16*)(ws + OFF_WXHT);
  _Float16* WhhT  = (_Float16*)(ws + OFF_WHHT);
  _Float16* WoutT = (_Float16*)(ws + OFF_WOUTT);
  float*    xp    = (float*)   (ws + OFF_XP);
  _Float16* z     = (_Float16*)(ws + OFF_Z);
  _Float16* hbuf0 = (_Float16*)(ws + OFF_H0);
  _Float16* hbuf1 = (_Float16*)(ws + OFF_H1);

  float* y     = (float*)d_out;                  // BT*OO
  float* hlast = y + (size_t)BT * OO;            // BB*HH

  // --- conversions / transposes ---
  {
    int n = BT * II;
    k_f32_to_f16<<<(n + 255) / 256, 256, 0, stream>>>(x, xf16, n);
  }
  {
    int n = BB * HH;  // initial hidden state -> f16 ping buffer
    k_f32_to_f16<<<(n + 255) / 256, 256, 0, stream>>>(h0, hbuf0, n);
  }
  {
    int n = II * HH;  // Wxh (rows 0..I-1 of W_rnn) -> (H x I)
    k_transpose_f16<<<(n + 255) / 256, 256, 0, stream>>>(W_rnn, WxhT, II, HH);
  }
  {
    int n = HH * HH;  // Whh (rows I.. of W_rnn) -> (H x H)
    k_transpose_f16<<<(n + 255) / 256, 256, 0, stream>>>(W_rnn + (size_t)II * HH, WhhT, HH, HH);
  }
  {
    int n = HH * OO;  // W_out -> (O x H)
    k_transpose_f16<<<(n + 255) / 256, 256, 0, stream>>>(W_out, WoutT, HH, OO);
  }

  // --- phase 1: xp (1024x64 tiles -> 65536 waves, 4 waves/block) ---
  k_gemm_xp<<<(BT / 16) * (HH / 16) / 4, 128, 0, stream>>>(xf16, WxhT, b_rnn, xp);

  // --- phase 2: 512 dependent steps; stream order is the global barrier ---
  for (int t = 0; t < TT; ++t) {
    const _Float16* hp = (t & 1) ? hbuf1 : hbuf0;
    _Float16*       hn = (t & 1) ? hbuf0 : hbuf1;
    k_recur<<<32, 128, 0, stream>>>(hp, WhhT, xp, hn, z, t);   // 128 waves
  }

  // --- phase 3: y (16384 waves) ---
  k_gemm_out<<<(BT / 16) * (OO / 16) / 4, 128, 0, stream>>>(z, WoutT, b_out, y);

  // --- h_last: step 511 wrote hbuf0 ---
  {
    int n = BB * HH;
    k_f16_to_f32<<<(n + 255) / 256, 256, 0, stream>>>(hbuf0, hlast, n);
  }
}
